// RNN_25709674233936
// MI455X (gfx1250) — compile-verified, matrix operand order checked by reading, not run on hardware
//
#include <hip/hip_runtime.h>
#include <hip/hip_bf16.h>

#define H         512
#define BSZ       256
#define TSTEPS    512
#define NIN       105
#define RULESTART 85
#define NRULE     20
#define NOUT      33
#define ALPHA_C   0.2f
#define SIGMA_C   0.05f

// padded LDS row strides (elements) to avoid bank conflicts on A-fragment reads
#define HS 520   // 512 + 8  -> row stride 1040B = 260 dwords, %64 = 4 banks rotation
#define XS 136   // 128 + 8

// fragment-layout weight tensors in workspace (elements of bf16/u16)
#define REC_ELEMS (32 * 16 * 512)  // [nt][kt][lane][16]
#define CAT_ELEMS (32 * 4 * 512)
#define OUT_ELEMS (3 * 16 * 512)

// dynamic-LDS carve (elements of u16); every sub-buffer 16B-aligned
#define SM_ELEMS (CAT_ELEMS + OUT_ELEMS + 16 * HS + 16 * XS)
#define SM_BYTES (SM_ELEMS * 2)

typedef __attribute__((ext_vector_type(16))) __bf16 v16bf;
typedef __attribute__((ext_vector_type(8)))  float  v8f;

union Frag { v16bf v; uint4 q[2]; };

__device__ __forceinline__ unsigned short f2bf(float f) {
  unsigned int u = __float_as_uint(f);
  unsigned int r = u + 0x7FFFu + ((u >> 16) & 1u);  // round-to-nearest-even
  return (unsigned short)(r >> 16);
}

// low 32 bits of a flat shared pointer == LDS byte offset (ISA §10.2 aperture)
__device__ __forceinline__ unsigned lds_off(const void* p) {
  return (unsigned)(unsigned long long)p;
}

// ASYNCcnt-tracked DMA: 16B global -> LDS, no VGPR round-trip (GVS addressing)
__device__ __forceinline__ void async_ld_b128(unsigned lds_byte, const void* gbase,
                                              unsigned goff) {
  asm volatile("global_load_async_to_lds_b128 %0, %1, %2"
               :: "v"(lds_byte), "v"(goff), "s"(gbase)
               : "memory");
}

// ---------------------------------------------------------------------------
// Prep: build bf16 weight tensors pre-permuted into the WMMA B-fragment
// layout (16-bit B 32x16: lane = 16*khalf + ncol; elem j -> K = 16*(j>>3) +
// 8*khalf + (j&7)).  B[k][n] = W[n][k] since all GEMMs are  act @ W^T.
// ---------------------------------------------------------------------------
__global__ void rnn_prep_kernel(const float* __restrict__ W_sensory,
                                const float* __restrict__ W_rule,
                                const float* __restrict__ W_rec,
                                const float* __restrict__ mask,
                                const float* __restrict__ W_out,
                                unsigned short* __restrict__ ws) {
  int id = blockIdx.x * blockDim.x + threadIdx.x;
  unsigned short* WBrec = ws;
  unsigned short* WBcat = ws + REC_ELEMS;
  unsigned short* WBout = ws + REC_ELEMS + CAT_ELEMS;

  if (id < REC_ELEMS) {
    int nt = id >> 13;  int rem = id & 8191;          // 16 kt * 512
    int kt = rem >> 9;  int le  = rem & 511;
    int lane = le >> 4; int j   = le & 15;
    int n  = nt * 16 + (lane & 15);
    int kk = ((j >> 3) << 4) + ((lane >> 4) << 3) + (j & 7);
    int k  = kt * 32 + kk;
    WBrec[id] = f2bf(W_rec[n * H + k] * mask[n * H + k]);
  } else if (id < REC_ELEMS + CAT_ELEMS) {
    int e = id - REC_ELEMS;
    int nt = e >> 11;   int rem = e & 2047;           // 4 kt * 512
    int kt = rem >> 9;  int le  = rem & 511;
    int lane = le >> 4; int j   = le & 15;
    int n  = nt * 16 + (lane & 15);
    int kk = ((j >> 3) << 4) + ((lane >> 4) << 3) + (j & 7);
    int k  = kt * 32 + kk;
    float v = 0.f;
    if (k < RULESTART)  v = W_sensory[n * RULESTART + k];
    else if (k < NIN)   v = W_rule[n * NRULE + (k - RULESTART)];
    WBcat[e] = f2bf(v);
  } else if (id < REC_ELEMS + CAT_ELEMS + OUT_ELEMS) {
    int e = id - REC_ELEMS - CAT_ELEMS;
    int nt = e >> 13;   int rem = e & 8191;
    int kt = rem >> 9;  int le  = rem & 511;
    int lane = le >> 4; int j   = le & 15;
    int n  = nt * 16 + (lane & 15);
    int kk = ((j >> 3) << 4) + ((lane >> 4) << 3) + (j & 7);
    int k  = kt * 32 + kk;
    WBout[e] = f2bf((n < NOUT) ? W_out[n * H + k] : 0.f);
  }
}

// ---------------------------------------------------------------------------
// Main fused RNN: 16 persistent WGs, each owns 16 batch rows for all T steps.
// h lives in VGPRs (fp32, C/D layout) + LDS (bf16, row-major) for A-frags.
// W_cat / W_out fragments async-DMA'd into LDS once; masked W_rec (512 KB)
// streams from L2 every step straight into VGPR B-fragments.
// ---------------------------------------------------------------------------
__global__ __launch_bounds__(512, 1)
void rnn_scan_kernel(const float* __restrict__ x,
                     const float* __restrict__ noise,
                     const float* __restrict__ b_sensory,
                     const float* __restrict__ b_rec,
                     const float* __restrict__ b_out,
                     const unsigned short* __restrict__ ws,
                     float* __restrict__ out) {
  const unsigned short* WBrec = ws;
  const unsigned short* WBcat = ws + REC_ELEMS;
  const unsigned short* WBout = ws + REC_ELEMS + CAT_ELEMS;

  extern __shared__ unsigned short smem[];
  unsigned short* catb = smem;                       // 131072 B
  unsigned short* outb = smem + CAT_ELEMS;           //  49152 B
  unsigned short* hbuf = smem + CAT_ELEMS + OUT_ELEMS;
  unsigned short* xbuf = hbuf + 16 * HS;

  const int tid   = threadIdx.x;
  const int lane  = tid & 31;
  const int wv    = tid >> 5;          // 16 waves
  const int khalf = lane >> 4;
  const int lm    = lane & 15;
  const int b0    = blockIdx.x * 16;

  // ---- one-time: async-DMA projection + readout B-fragments into LDS ------
  {
    unsigned lbase = lds_off(catb);
    for (unsigned ofs = (unsigned)tid * 16u; ofs < CAT_ELEMS * 2u; ofs += 512u * 16u)
      async_ld_b128(lbase + ofs, WBcat, ofs);
    lbase = lds_off(outb);
    for (unsigned ofs = (unsigned)tid * 16u; ofs < OUT_ELEMS * 2u; ofs += 512u * 16u)
      async_ld_b128(lbase + ofs, WBout, ofs);
  }

  for (int i = tid; i < 16 * HS; i += 512) hbuf[i] = 0;
  for (int i = tid; i < 16 * XS; i += 512) xbuf[i] = 0;

  const int nt0 = wv * 2, nt1 = wv * 2 + 1;
  const int n0 = nt0 * 16 + lm, n1 = nt1 * 16 + lm;
  const float bias0 = b_sensory[n0] + b_rec[n0];
  const float bias1 = b_sensory[n1] + b_rec[n1];

  int   n_out  = 0;
  float bout_n = 0.f;
  if (wv >= 13) { n_out = (wv - 13) * 16 + lm; if (n_out < NOUT) bout_n = b_out[n_out]; }

  v8f h0 = {}, h1 = {};   // persistent hidden state, fp32, D layout

  asm volatile("s_wait_asynccnt 0" ::: "memory");  // my DMA writes landed
  __syncthreads();                                  // everyone's landed

  for (int t = 0; t < TSTEPS; ++t) {
    // ---- phase A: stage x[t] tile into LDS as bf16; prefetch noise -------
    for (int i = tid; i < 16 * NIN; i += 512) {
      int m = i / NIN, k = i - m * NIN;
      xbuf[m * XS + k] = f2bf(x[(t * BSZ + b0 + m) * NIN + k]);
    }
    float nz0[8], nz1[8];
    #pragma unroll
    for (int v = 0; v < 8; ++v) {
      int m = v + 8 * khalf;
      int base = (t * BSZ + b0 + m) * H;
      nz0[v] = noise[base + n0];
      nz1[v] = noise[base + n1];
    }
    if (t + 1 < TSTEPS && lane < 7) {  // warm caches for next x tile
      const float* nx = &x[((t + 1) * BSZ + b0) * NIN];
      __builtin_prefetch(nx + (wv * 7 + lane) * 15, 0, 1);
    }
    __syncthreads();

    // ---- phase B: pre = x@Wcat^T + h@(Wrec.mask)^T via bf16 WMMA ----------
    v8f acc0 = {}, acc1 = {};
    #pragma unroll
    for (int kt = 0; kt < 4; ++kt) {                  // projection, K = 128
      Frag a;
      const unsigned short* r = &xbuf[lm * XS + kt * 32 + khalf * 8];
      a.q[0] = *(const uint4*)r;
      a.q[1] = *(const uint4*)(r + 16);
      Frag bA, bB;
      const uint4* p0 = (const uint4*)&catb[((nt0 * 4 + kt) * 32 + lane) * 16];
      const uint4* p1 = (const uint4*)&catb[((nt1 * 4 + kt) * 32 + lane) * 16];
      bA.q[0] = p0[0]; bA.q[1] = p0[1];
      bB.q[0] = p1[0]; bB.q[1] = p1[1];
      acc0 = __builtin_amdgcn_wmma_f32_16x16x32_bf16(false, a.v, false, bA.v, (short)0, acc0, false, false);
      acc1 = __builtin_amdgcn_wmma_f32_16x16x32_bf16(false, a.v, false, bB.v, (short)0, acc1, false, false);
    }
    #pragma unroll 8
    for (int kt = 0; kt < 16; ++kt) {                 // recurrence, K = 512
      Frag a;
      const unsigned short* r = &hbuf[lm * HS + kt * 32 + khalf * 8];
      a.q[0] = *(const uint4*)r;
      a.q[1] = *(const uint4*)(r + 16);
      Frag bA, bB;
      const uint4* p0 = (const uint4*)(WBrec + (size_t)((nt0 * 16 + kt) * 32 + lane) * 16);
      const uint4* p1 = (const uint4*)(WBrec + (size_t)((nt1 * 16 + kt) * 32 + lane) * 16);
      bA.q[0] = p0[0]; bA.q[1] = p0[1];
      bB.q[0] = p1[0]; bB.q[1] = p1[1];
      acc0 = __builtin_amdgcn_wmma_f32_16x16x32_bf16(false, a.v, false, bA.v, (short)0, acc0, false, false);
      acc1 = __builtin_amdgcn_wmma_f32_16x16x32_bf16(false, a.v, false, bB.v, (short)0, acc1, false, false);
    }
    // ---- leaky ReLU integration (fp32, in registers) ----------------------
    #pragma unroll
    for (int v = 0; v < 8; ++v) {
      float p0 = acc0[v] + SIGMA_C * nz0[v] + bias0;
      float p1 = acc1[v] + SIGMA_C * nz1[v] + bias1;
      p0 = fmaxf(p0, 0.f);
      p1 = fmaxf(p1, 0.f);
      h0[v] = ALPHA_C * p0 + (1.f - ALPHA_C) * h0[v];
      h1[v] = ALPHA_C * p1 + (1.f - ALPHA_C) * h1[v];
    }
    __syncthreads();   // everyone done reading old hbuf

    // ---- phase C: publish h_new to LDS as bf16 ----------------------------
    #pragma unroll
    for (int v = 0; v < 8; ++v) {
      int m = v + 8 * khalf;
      hbuf[m * HS + n0] = f2bf(h0[v]);
      hbuf[m * HS + n1] = f2bf(h1[v]);
    }
    __syncthreads();

    // ---- phase D: readout  out[t] = h @ W_out^T + b_out (waves 13..15) ----
    if (wv >= 13) {
      const int ot = wv - 13;
      v8f oacc = {};
      #pragma unroll 4
      for (int kt = 0; kt < 16; ++kt) {
        Frag a;
        const unsigned short* r = &hbuf[lm * HS + kt * 32 + khalf * 8];
        a.q[0] = *(const uint4*)r;
        a.q[1] = *(const uint4*)(r + 16);
        Frag bo;
        const uint4* p = (const uint4*)&outb[((ot * 16 + kt) * 32 + lane) * 16];
        bo.q[0] = p[0]; bo.q[1] = p[1];
        oacc = __builtin_amdgcn_wmma_f32_16x16x32_bf16(false, a.v, false, bo.v, (short)0, oacc, false, false);
      }
      if (n_out < NOUT) {
        #pragma unroll
        for (int v = 0; v < 8; ++v) {
          int m = v + 8 * khalf;
          out[(t * BSZ + b0 + m) * NOUT + n_out] = oacc[v] + bout_n;
        }
      }
    }
  }
}

extern "C" void kernel_launch(void* const* d_in, const int* in_sizes, int n_in,
                              void* d_out, int out_size, void* d_ws, size_t ws_size,
                              hipStream_t stream) {
  const float* x         = (const float*)d_in[0];
  const float* noise     = (const float*)d_in[1];
  const float* mask      = (const float*)d_in[2];
  const float* W_sensory = (const float*)d_in[3];
  const float* b_sensory = (const float*)d_in[4];
  const float* W_rule    = (const float*)d_in[5];
  const float* W_rec     = (const float*)d_in[6];
  const float* b_rec     = (const float*)d_in[7];
  const float* W_out     = (const float*)d_in[8];
  const float* b_out     = (const float*)d_in[9];
  float* out = (float*)d_out;
  unsigned short* ws = (unsigned short*)d_ws;

  (void)in_sizes; (void)n_in; (void)out_size; (void)ws_size;

  const int total = REC_ELEMS + CAT_ELEMS + OUT_ELEMS;
  rnn_prep_kernel<<<(total + 255) / 256, 256, 0, stream>>>(
      W_sensory, W_rule, W_rec, mask, W_out, ws);
  rnn_scan_kernel<<<BSZ / 16, 512, SM_BYTES, stream>>>(
      x, noise, b_sensory, b_rec, b_out, ws, out);
}
